// AKConv_22488448762350
// MI455X (gfx1250) — compile-verified, matrix operand order checked by reading, not run on hardware
//
#include <hip/hip_runtime.h>
#include <math.h>

// ---------------------------------------------------------------------------
// AKConv for MI455X (gfx1250, wave32, WMMA)
// shapes: B=8, INC=128, OUTC=256, N=9, H=W=64, STRIDE=1
// y shape: (8, 256, 64, 576) -> 75,497,472 fp32 elements
// ---------------------------------------------------------------------------

typedef __attribute__((ext_vector_type(16))) _Float16 v16h;
typedef __attribute__((ext_vector_type(8)))  float    v8f;

#define AK_B     8
#define AK_INC   128
#define AK_OUTC  256
#define AK_N     9
#define AK_H     64
#define AK_W     64
#define AK_HW    4096          // 64*64
#define AK_POS   36864         // 64*64*9  positions per batch
#define AK_CNT   294912.0f     // 8*64*576 elements per channel for BN

// 16-bit A/B fragment K mapping for v_wmma_f32_16x16x32_f16 (wave32):
// lanes 0-15:  VGPR0..3 -> K 0..7,  VGPR4..7 -> K 16..23
// lanes 16-31: VGPR0..3 -> K 8..15, VGPR4..7 -> K 24..31
__device__ __forceinline__ int ak_kmap(int i, int h) {
    return (i < 8) ? (h * 8 + i) : (16 + h * 8 + (i - 8));
}

// ---------------------------------------------------------------------------
// Kernel 1: 3x3 offset conv, 128 -> 18 channels, zero padding. x is L2-resident.
// one thread per output element; 589,824 threads total.
// ---------------------------------------------------------------------------
__global__ __launch_bounds__(256) void ak_offset_conv(
    const float* __restrict__ x, const float* __restrict__ pw,
    const float* __restrict__ pb, float* __restrict__ offs)
{
    int idx = blockIdx.x * 256 + threadIdx.x;        // ((b*18+j)*64+hh)*64+ww
    int ww = idx & 63;
    int hh = (idx >> 6) & 63;
    int bj = idx >> 12;
    int j  = bj % 18;
    int bb = bj / 18;

    float acc = pb[j];
    const float* xb0 = x  + ((size_t)bb * AK_INC) * AK_HW;
    const float* wj0 = pw + (size_t)j * AK_INC * 9;
    for (int c = 0; c < AK_INC; ++c) {
        const float* xb = xb0 + (size_t)c * AK_HW;
        const float* wj = wj0 + c * 9;
#pragma unroll
        for (int ky = 0; ky < 3; ++ky) {
            int yy = hh + ky - 1;
            if (yy < 0 || yy > 63) continue;
#pragma unroll
            for (int kx = 0; kx < 3; ++kx) {
                int xx = ww + kx - 1;
                if (xx < 0 || xx > 63) continue;
                acc = fmaf(xb[yy * 64 + xx], wj[ky * 3 + kx], acc);
            }
        }
    }
    offs[idx] = acc;
}

// ---------------------------------------------------------------------------
// Kernel 2: zero the BN stats accumulators (sum[256], sumsq[256]) in ws.
// ---------------------------------------------------------------------------
__global__ void ak_zero_stats(float* __restrict__ stats)
{
    if (threadIdx.x < 512) stats[threadIdx.x] = 0.0f;
}

// ---------------------------------------------------------------------------
// Kernel 3: fused deformable-gather + 1x1-conv GEMM via WMMA.
// grid = (576, 8): blockIdx.x -> 64-position tile, blockIdx.y -> batch.
// block = 256 threads = 8 waves. Each WG: 64 pos x 256 outch.
// ---------------------------------------------------------------------------
__global__ __launch_bounds__(256) void ak_gemm(
    const float* __restrict__ x, const float* __restrict__ offs,
    const float* __restrict__ cw, float* __restrict__ y)
{
    __shared__ _Float16 s_xoff[64 * AK_INC];   // [pos_local][c], 16 KB
    __shared__ int      s_idx[64][4];          // bilinear gather indices
    __shared__ float    s_w[64][4];            // bilinear weights

    const int tid     = threadIdx.x;
    const int bb      = blockIdx.y;
    const int posBase = blockIdx.x * 64;

    // --- Phase A: per-position offset points, indices, weights (64 threads)
    if (tid < 64) {
        int pos = posBase + tid;               // (hh*64+ww)*9 + nn
        int nn  = pos % 9;
        int sp  = pos / 9;
        int ww  = sp & 63;
        int hh  = sp >> 6;
        float ang = 6.28318530717958647692f * (float)nn / 9.0f;
        const float* ob = offs + ((size_t)bb * 18) * AK_HW;
        float px = ob[(size_t)nn * AK_HW + hh * 64 + ww] + __cosf(ang) + (float)hh;
        float py = ob[(size_t)(9 + nn) * AK_HW + hh * 64 + ww] + __sinf(ang) + (float)ww;
        float fx = floorf(px), fy = floorf(py);
        float qltx = fminf(fmaxf(fx,        0.f), 63.f);
        float qlty = fminf(fmaxf(fy,        0.f), 63.f);
        float qrbx = fminf(fmaxf(fx + 1.f,  0.f), 63.f);
        float qrby = fminf(fmaxf(fy + 1.f,  0.f), 63.f);
        float pxc  = fminf(fmaxf(px,        0.f), 63.f);
        float pyc  = fminf(fmaxf(py,        0.f), 63.f);
        s_w[tid][0] = (1.f + (qltx - pxc)) * (1.f + (qlty - pyc)); // lt
        s_w[tid][1] = (1.f - (qrbx - pxc)) * (1.f - (qrby - pyc)); // rb
        s_w[tid][2] = (1.f + (qltx - pxc)) * (1.f - (qrby - pyc)); // lb
        s_w[tid][3] = (1.f - (qrbx - pxc)) * (1.f + (qlty - pyc)); // rt
        s_idx[tid][0] = (int)qltx * 64 + (int)qlty;
        s_idx[tid][1] = (int)qrbx * 64 + (int)qrby;
        s_idx[tid][2] = (int)qltx * 64 + (int)qrby;
        s_idx[tid][3] = (int)qrbx * 64 + (int)qlty;
    }
    __syncthreads();

    // --- Phase B: gather/blend x -> f16 x_off tile in LDS ([pos][c])
    {
        const int pl = tid >> 2;               // pos_local 0..63
        const int c0 = tid & 3;                // c quarter 0..3
        const int i0 = s_idx[pl][0], i1 = s_idx[pl][1];
        const int i2 = s_idx[pl][2], i3 = s_idx[pl][3];
        const float w0 = s_w[pl][0], w1 = s_w[pl][1];
        const float w2 = s_w[pl][2], w3 = s_w[pl][3];
        const float* xbase = x + ((size_t)bb * AK_INC) * AK_HW;
#pragma unroll 4
        for (int k = 0; k < 32; ++k) {
            int c = c0 * 32 + k;
            const float* xb = xbase + (size_t)c * AK_HW;
            float v = w0 * xb[i0] + w1 * xb[i1] + w2 * xb[i2] + w3 * xb[i3];
            s_xoff[pl * AK_INC + c] = (_Float16)v;
        }
    }
    __syncthreads();

    // --- Phase C: WMMA  y[16o x 16pos] += cw[16o x 32c] * xoff[32c x 16pos]
    const int lane = tid & 31;
    const int half = lane >> 4;
    const int m    = lane & 15;
    const int wv   = tid >> 5;                 // wave 0..7

    // preload A fragments (cw as f16): wave wv owns out-tiles {2wv, 2wv+1}
    v16h afrag[2][4];
#pragma unroll
    for (int oo = 0; oo < 2; ++oo) {
        int orow = (wv * 2 + oo) * 16 + m;
#pragma unroll
        for (int ks = 0; ks < 4; ++ks) {
            const float* cwp = cw + (size_t)orow * AK_INC + ks * 32;
#pragma unroll
            for (int i = 0; i < 16; ++i)
                afrag[oo][ks][i] = (_Float16)cwp[ak_kmap(i, half)];
        }
    }

#pragma unroll
    for (int tt = 0; tt < 8; ++tt) {
        int oo = tt >> 2;
        int pt = tt & 3;
        v8f acc = {};
#pragma unroll
        for (int ks = 0; ks < 4; ++ks) {
            v16h bfrag;
            const _Float16* xo = &s_xoff[(pt * 16 + m) * AK_INC + ks * 32];
#pragma unroll
            for (int i = 0; i < 16; ++i) bfrag[i] = xo[ak_kmap(i, half)];
            acc = __builtin_amdgcn_wmma_f32_16x16x32_f16(
                /*neg_a=*/false, afrag[oo][ks],
                /*neg_b=*/false, bfrag,
                /*c_mod=*/(short)0, acc,
                /*reuse_a=*/false, /*reuse_b=*/false);
        }
        // store D: lane 0-15 -> rows r, lanes 16-31 -> rows r+8; col = pos
        int otile = wv * 2 + oo;
        int posG  = posBase + pt * 16 + m;
#pragma unroll
        for (int r = 0; r < 8; ++r) {
            int o = otile * 16 + r + half * 8;
            y[((size_t)bb * AK_OUTC + o) * AK_POS + posG] = acc[r];
        }
    }
}

// ---------------------------------------------------------------------------
// Kernel 4: BN statistics, float4 loads (bandwidth-bound pass).
// grid = 2048 blocks: block -> (channel o, batch b). 36864/4 = 9216 float4.
// ---------------------------------------------------------------------------
__global__ __launch_bounds__(256) void ak_stats(
    const float* __restrict__ y, float* __restrict__ stats)
{
    __shared__ float ss[256];
    __shared__ float sq[256];
    int o  = blockIdx.x >> 3;
    int bb = blockIdx.x & 7;
    const float4* yb = (const float4*)(y + ((size_t)bb * AK_OUTC + o) * AK_POS);
    float s = 0.f, q = 0.f;
#pragma unroll 4
    for (int i = threadIdx.x; i < AK_POS / 4; i += 256) {
        float4 v = yb[i];
        s += v.x + v.y + v.z + v.w;
        q += v.x * v.x + v.y * v.y + v.z * v.z + v.w * v.w;
    }
    ss[threadIdx.x] = s;
    sq[threadIdx.x] = q;
    __syncthreads();
    for (int st = 128; st > 0; st >>= 1) {
        if (threadIdx.x < st) {
            ss[threadIdx.x] += ss[threadIdx.x + st];
            sq[threadIdx.x] += sq[threadIdx.x + st];
        }
        __syncthreads();
    }
    if (threadIdx.x == 0) {
        atomicAdd(&stats[o],       ss[0]);
        atomicAdd(&stats[256 + o], sq[0]);
    }
}

// ---------------------------------------------------------------------------
// Kernel 5: in-place BatchNorm + SiLU, float4 RMW (bandwidth-bound pass).
// AK_POS % 4 == 0, so a float4 never straddles a channel boundary.
// grid = 75,497,472 / (256*4) = 73728 blocks.
// ---------------------------------------------------------------------------
__global__ __launch_bounds__(256) void ak_norm(
    float* __restrict__ y, const float* __restrict__ stats,
    const float* __restrict__ gamma, const float* __restrict__ beta)
{
    size_t idx4 = (size_t)blockIdx.x * 256 + threadIdx.x;   // float4 index
    int o = (int)(((idx4 * 4) / AK_POS) & 255);
    float mean = stats[o] * (1.0f / AK_CNT);
    float var  = stats[256 + o] * (1.0f / AK_CNT) - mean * mean;
    float inv  = rsqrtf(var + 1e-5f) * gamma[o];
    float bta  = beta[o];

    float4* yp = (float4*)y + idx4;
    float4 v = *yp;
    float a0 = (v.x - mean) * inv + bta;
    float a1 = (v.y - mean) * inv + bta;
    float a2 = (v.z - mean) * inv + bta;
    float a3 = (v.w - mean) * inv + bta;
    v.x = a0 / (1.0f + __expf(-a0));
    v.y = a1 / (1.0f + __expf(-a1));
    v.z = a2 / (1.0f + __expf(-a2));
    v.w = a3 / (1.0f + __expf(-a3));
    *yp = v;
}

// ---------------------------------------------------------------------------
// Host launcher
// ---------------------------------------------------------------------------
extern "C" void kernel_launch(void* const* d_in, const int* in_sizes, int n_in,
                              void* d_out, int out_size, void* d_ws, size_t ws_size,
                              hipStream_t stream)
{
    (void)in_sizes; (void)n_in; (void)out_size; (void)ws_size;
    const float* x     = (const float*)d_in[0];   // (8,128,64,64)
    const float* pw    = (const float*)d_in[1];   // (18,128,3,3)
    const float* pb    = (const float*)d_in[2];   // (18,)
    const float* cw    = (const float*)d_in[3];   // (256,128,1,1)
    const float* gamma = (const float*)d_in[4];   // (256,)
    const float* beta  = (const float*)d_in[5];   // (256,)
    float*       out   = (float*)d_out;           // (8,256,64,576)

    float* offs  = (float*)d_ws;                  // 8*18*64*64 floats = 2.36 MB
    float* stats = offs + AK_B * 18 * AK_HW;      // sum[256] + sumsq[256]

    // 1) offset conv: 589824 outputs
    ak_offset_conv<<<2304, 256, 0, stream>>>(x, pw, pb, offs);
    // 2) zero BN accumulators
    ak_zero_stats<<<1, 512, 0, stream>>>(stats);
    // 3) fused gather + WMMA GEMM: (pos tiles = 36864/64 = 576) x (batch = 8)
    ak_gemm<<<dim3(576, 8), 256, 0, stream>>>(x, offs, cw, out);
    // 4) BN stats: 256 channels x 8 batch-chunks, float4 loads
    ak_stats<<<2048, 256, 0, stream>>>(out, stats);
    // 5) BN + SiLU in place: float4 RMW
    ak_norm<<<73728, 256, 0, stream>>>(out, stats, gamma, beta);
}
